// SOMNetwork_79637283602876
// MI455X (gfx1250) — compile-verified, matrix operand order checked by reading, not run on hardware
//
#include <hip/hip_runtime.h>
#include <hip/hip_bf16.h>
#include <math.h>

typedef _Float16 v16h __attribute__((ext_vector_type(16)));
typedef float    v8f  __attribute__((ext_vector_type(8)));

#define WMMA_F16(a, b, c) \
  __builtin_amdgcn_wmma_f32_16x16x32_f16(false, (a), false, (b), (short)0, (c), false, false)

// ---------------------------------------------------------------------------
// Load one 16x32 WMMA operand tile (A or B) from a row-major [nrows x ld]
// fp32 matrix, converting to f16 and accumulating the fp32 sum of squares of
// this lane's elements (for the ||.||^2 terms of the distance GEMM).
// Layout (ISA 7.12.2, 16-bit operands, wave32):
//   lane < 16 : row = lane,      elements 0..7 -> K{0..7},  8..15 -> K{16..23}
//   lane >= 16: row = lane - 16, elements 0..7 -> K{8..15}, 8..15 -> K{24..31}
// ---------------------------------------------------------------------------
__device__ __forceinline__ v16h load_rowK(const float* __restrict__ src, int ld,
                                          int row0, int nrows, int K, int kbase,
                                          int lane, float& sq) {
  const int h = lane >> 4;
  const int r = lane & 15;
  const int grow = row0 + r;
  const int rok = (grow < nrows);
  const float* rp = src + (size_t)(rok ? grow : 0) * (size_t)ld;
  v16h out;
#pragma unroll
  for (int i = 0; i < 16; ++i) {
    int k = kbase + (i & 7) + (h << 3) + ((i >> 3) << 4);
    int kc = (k < K) ? k : 0;
    float v = rp[kc];
    v = (rok && (k < K)) ? v : 0.0f;
    sq += v * v;
    out[i] = (_Float16)v;
  }
  return out;
}

// ---------------------------------------------------------------------------
// Stage 1: dist1[b, o, p] = sum_c sqrt(max(||patch||^2+||w||^2-2*dot, 1e-12))
// x: [4096,3,28,28], w: [100,3,5,5], stride 4, 6x6 output pixels (p = ph*6+pw)
// One wave computes a 16(b) x 16(o) tile for one pixel; 3 WMMAs (one per ch).
// Fused: per-wave partial (sum, sum^2) of the valid dist entries for pixel p.
// ---------------------------------------------------------------------------
__global__ void __launch_bounds__(32)
stage1_dist_kernel(const float* __restrict__ x, const float* __restrict__ w,
                   float* __restrict__ dist1, float* __restrict__ partials) {
  const int lane = threadIdx.x;
  const int m0 = blockIdx.x << 4;   // batch tile (4096/16 = 256)
  const int n0 = blockIdx.y << 4;   // filter tile (ceil(100/16) = 7)
  const int p  = blockIdx.z;        // pixel 0..35
  const int ph = p / 6, pw = p % 6;
  const int h = lane >> 4, r = lane & 15;

  float dacc[8];
#pragma unroll
  for (int v = 0; v < 8; ++v) dacc[v] = 0.0f;

  for (int c = 0; c < 3; ++c) {
    float pn = 0.0f, wn = 0.0f;
    // A tile: 5x5 window of x for 16 batches (K = 25, padded to 32)
    const int b = m0 + r;  // always < 4096
    const float* base = x + ((((size_t)b * 3 + c) * 28 + (size_t)ph * 4) * 28 + pw * 4);
    v16h a;
#pragma unroll
    for (int i = 0; i < 16; ++i) {
      int k = (i & 7) + (h << 3) + ((i >> 3) << 4);
      int kc = (k < 25) ? k : 0;
      float v = base[(kc / 5) * 28 + (kc % 5)];
      v = (k < 25) ? v : 0.0f;
      pn += v * v;
      a[i] = (_Float16)v;
    }
    // B tile: filter rows for channel c (ld = 75, 25 taps)
    v16h bm = load_rowK(w + c * 25, 75, n0, 100, 25, 0, lane, wn);
    v8f acc = {};
    acc = WMMA_F16(a, bm, acc);
    // combine the split-lane halves of the squared norms
    pn += __shfl_xor(pn, 16, 32);   // lane l now holds ||row (l&15)||^2
    wn += __shfl_xor(wn, 16, 32);   // lane l now holds ||filter n0+(l&15)||^2
#pragma unroll
    for (int v = 0; v < 8; ++v) {
      int Mrow = v + (h << 3);                  // D: M = v + 8*(lane>=16)
      float pnM = __shfl(pn, Mrow, 32);
      float d2 = pnM + wn - 2.0f * acc[v];
      d2 = fmaxf(d2, 1e-12f);
      dacc[v] += sqrtf(d2);
    }
  }
  // store + fused per-pixel stats partial
  float s = 0.0f, s2 = 0.0f;
#pragma unroll
  for (int v = 0; v < 8; ++v) {
    int gm = m0 + v + (h << 3);
    int gn = n0 + r;
    if (gn < 100) {
      float d = dacc[v];
      dist1[((size_t)gm * 100 + gn) * 36 + p] = d;
      s += d; s2 += d * d;
    }
  }
#pragma unroll
  for (int o = 16; o > 0; o >>= 1) { s += __shfl_xor(s, o, 32); s2 += __shfl_xor(s2, o, 32); }
  if (lane == 0) {
    size_t flat = (size_t)blockIdx.y * gridDim.x + blockIdx.x;   // 0..1791
    partials[((size_t)p * 1792 + flat) * 2]     = s;
    partials[((size_t)p * 1792 + flat) * 2 + 1] = s2;
  }
}

// ---------------------------------------------------------------------------
// Generic distance GEMM for stages 2-4 (single channel, single pixel):
// dist[m,n] = sqrt(max(||A_m||^2 + ||W_n||^2 - 2*A_m.W_n, 1e-12))
// A: [M x K], W: [N x K] row-major. One wave computes a 16(M) x 32(N) tile:
// two accumulators reuse the A operand; K-loop of paired 16x16x32 WMMAs.
// Fused per-wave (sum, sum^2) partial over the valid entries it wrote.
// ---------------------------------------------------------------------------
__global__ void __launch_bounds__(32)
rbf_dist_kernel(const float* __restrict__ A, const float* __restrict__ W,
                float* __restrict__ dist, float* __restrict__ partials,
                int M, int N, int K) {
  const int lane = threadIdx.x;
  const int m0 = blockIdx.x << 4;
  const int n0 = blockIdx.y << 5;         // 32-wide N tile
  const int h = lane >> 4, r = lane & 15;
  float pn = 0.0f, wn0 = 0.0f, wn1 = 0.0f;
  v8f acc0 = {}, acc1 = {};
  const int wrow = (n0 + r < N) ? (n0 + r) : (N - 1);
  for (int k0 = 0; k0 < K; k0 += 32) {
    if (k0 + 32 < K) {  // speculative prefetch of the next K-tile (RT hint)
      __builtin_prefetch(A + (size_t)(m0 + r) * K + k0 + 32, 0, 0);
      __builtin_prefetch(W + (size_t)wrow * K + k0 + 32, 0, 0);
    }
    v16h a  = load_rowK(A, K, m0, M, K, k0, lane, pn);
    v16h b0 = load_rowK(W, K, n0,      N, K, k0, lane, wn0);
    v16h b1 = load_rowK(W, K, n0 + 16, N, K, k0, lane, wn1);
    acc0 = WMMA_F16(a, b0, acc0);
    acc1 = WMMA_F16(a, b1, acc1);
  }
  pn  += __shfl_xor(pn, 16, 32);
  wn0 += __shfl_xor(wn0, 16, 32);
  wn1 += __shfl_xor(wn1, 16, 32);
  float s = 0.0f, s2 = 0.0f;
#pragma unroll
  for (int v = 0; v < 8; ++v) {
    int Mrow = v + (h << 3);
    float pnM = __shfl(pn, Mrow, 32);
    int gm = m0 + Mrow;
    {
      float d = sqrtf(fmaxf(pnM + wn0 - 2.0f * acc0[v], 1e-12f));
      int gn = n0 + r;
      if (gm < M && gn < N) { dist[(size_t)gm * N + gn] = d; s += d; s2 += d * d; }
    }
    {
      float d = sqrtf(fmaxf(pnM + wn1 - 2.0f * acc1[v], 1e-12f));
      int gn = n0 + 16 + r;
      if (gm < M && gn < N) { dist[(size_t)gm * N + gn] = d; s += d; s2 += d * d; }
    }
  }
#pragma unroll
  for (int o = 16; o > 0; o >>= 1) { s += __shfl_xor(s, o, 32); s2 += __shfl_xor(s2, o, 32); }
  if (lane == 0) {
    size_t bi = (size_t)blockIdx.y * gridDim.x + blockIdx.x;
    partials[bi * 2]     = s;
    partials[bi * 2 + 1] = s2;
  }
}

// ---------------------------------------------------------------------------
// Deterministic fixed-order finalize of (sum, sum^2) partials.
// ---------------------------------------------------------------------------
__global__ void finalize_stage1_kernel(const float* __restrict__ partials,
                                       int nblk, float* __restrict__ stats) {
  int p = threadIdx.x;
  if (p >= 36) return;
  float s = 0.0f, s2 = 0.0f;
  for (int i = 0; i < nblk; ++i) {
    s  += partials[((size_t)p * nblk + i) * 2];
    s2 += partials[((size_t)p * nblk + i) * 2 + 1];
  }
  stats[p * 2] = s; stats[p * 2 + 1] = s2;
}

__global__ void finalize_pairs_kernel(const float* __restrict__ partials,
                                      int nblk, float* __restrict__ pair) {
  float s = 0.0f, s2 = 0.0f;
  for (int i = threadIdx.x; i < nblk; i += blockDim.x) {
    s += partials[(size_t)i * 2]; s2 += partials[(size_t)i * 2 + 1];
  }
#pragma unroll
  for (int o = 16; o > 0; o >>= 1) { s += __shfl_xor(s, o, 32); s2 += __shfl_xor(s2, o, 32); }
  __shared__ float sh[16];
  int lane = threadIdx.x & 31, wid = threadIdx.x >> 5;
  if (lane == 0) { sh[wid] = s; sh[8 + wid] = s2; }
  __syncthreads();
  if (threadIdx.x == 0) {
    float ts = 0.0f, t2 = 0.0f;
    int nw = blockDim.x >> 5;
    for (int w = 0; w < nw; ++w) { ts += sh[w]; t2 += sh[8 + w]; }
    pair[0] = ts; pair[1] = t2;
  }
}

__device__ __forceinline__ float std_from(const float* pair, float cnt) {
  float s = pair[0], s2 = pair[1];
  float var = (s2 - s * s / cnt) / (cnt - 1.0f);   // torch.std, ddof=1
  return sqrtf(fmaxf(var, 0.0f));
}

// ---------------------------------------------------------------------------
// Stage 1 apply + SFM1: out1 = crelu(exp(-d^2/(2 std_p^2)), 0.4) [B,100,6,6]
// -> alpha-weighted 2x2 mean pool -> images2 [B*9, 100]
// ---------------------------------------------------------------------------
__global__ void apply1_sfm1_kernel(const float* __restrict__ dist1,
                                   const float* __restrict__ stats,
                                   float* __restrict__ img2) {
  int idx = blockIdx.x * blockDim.x + threadIdx.x;
  if (idx >= 36864 * 100) return;
  int o = idx % 100;
  int img = idx / 100;          // b*9 + i*3 + j
  int b = img / 9, ij = img % 9, i = ij / 3, j = ij % 3;
  const float alpha[2][2] = {{0.729f, 0.81f}, {0.9f, 1.0f}};  // 0.9^[3,2;1,0]
  float acc = 0.0f;
#pragma unroll
  for (int di = 0; di < 2; ++di) {
#pragma unroll
    for (int dj = 0; dj < 2; ++dj) {
      int p = (2 * i + di) * 6 + (2 * j + dj);
      float sd = std_from(stats + 2 * p, 409600.0f);
      float d = dist1[((size_t)b * 100 + o) * 36 + p];
      float val = expf(-(d * d) / (2.0f * sd * sd));
      val = (val >= 0.4f) ? val : 0.0f;
      acc += alpha[di][dj] * val;
    }
  }
  img2[idx] = acc * 0.25f;
}

// Stage 2 apply + SFM2: [B*9,225] -> (alpha over j of 3x3 group) -> [B*3,225]
__global__ void apply2_sfm2_kernel(const float* __restrict__ dist2,
                                   const float* __restrict__ pair,
                                   float* __restrict__ img3) {
  int idx = blockIdx.x * blockDim.x + threadIdx.x;
  if (idx >= 12288 * 225) return;
  int o = idx % 225;
  int bi = idx / 225;           // b*3 + i
  int b = bi / 3, i = bi % 3;
  float sd = std_from(pair, 8294400.0f);
  float inv = 1.0f / (2.0f * sd * sd);
  const float a2[3] = {0.81f, 0.9f, 1.0f};   // 0.9^[2,1,0]
  float acc = 0.0f;
#pragma unroll
  for (int j = 0; j < 3; ++j) {
    float d = dist2[((size_t)(b * 9 + i * 3 + j)) * 225 + o];
    float v = expf(-(d * d) * inv);
    v = (v >= 0.1f) ? v : 0.0f;
    acc += a2[j] * v;
  }
  img3[idx] = acc * (1.0f / 3.0f);
}

// Stage 3 apply + SFM3: [B*3,625] -> (alpha over i) -> [B,625]
__global__ void apply3_sfm3_kernel(const float* __restrict__ dist3,
                                   const float* __restrict__ pair,
                                   float* __restrict__ img4) {
  int idx = blockIdx.x * blockDim.x + threadIdx.x;
  if (idx >= 4096 * 625) return;
  int o = idx % 625;
  int b = idx / 625;
  float sd = std_from(pair, 7680000.0f);
  float inv = 1.0f / (2.0f * sd * sd);
  const float a3[3] = {0.81f, 0.9f, 1.0f};
  float acc = 0.0f;
#pragma unroll
  for (int i = 0; i < 3; ++i) {
    float d = dist3[((size_t)(b * 3 + i)) * 625 + o];
    float v = expf(-(d * d) * inv);
    v = (v >= 0.01f) ? v : 0.0f;
    acc += a3[i] * v;
  }
  img4[idx] = acc * (1.0f / 3.0f);
}

// Stage 4 apply + FC: out[b,j] = fc_b[j] + sum_o crelu(exp(...),0.01)*fc_w[j,o]
__global__ void apply4_fc_kernel(const float* __restrict__ dist4,
                                 const float* __restrict__ pair,
                                 const float* __restrict__ fc_w,
                                 const float* __restrict__ fc_b,
                                 float* __restrict__ out) {
  int idx = blockIdx.x * blockDim.x + threadIdx.x;
  if (idx >= 4096 * 10) return;
  int b = idx / 10, jj = idx % 10;
  float sd = std_from(pair, 5017600.0f);
  float inv = 1.0f / (2.0f * sd * sd);
  float acc = fc_b[jj];
  const float* dr = dist4 + (size_t)b * 1225;
  const float* wr = fc_w + (size_t)jj * 1225;
  for (int o = 0; o < 1225; ++o) {
    float d = dr[o];
    float v = expf(-(d * d) * inv);
    v = (v >= 0.01f) ? v : 0.0f;
    acc += v * wr[o];
  }
  out[idx] = acc;
}

// ---------------------------------------------------------------------------
extern "C" void kernel_launch(void* const* d_in, const int* in_sizes, int n_in,
                              void* d_out, int out_size, void* d_ws, size_t ws_size,
                              hipStream_t stream) {
  (void)in_sizes; (void)n_in; (void)out_size; (void)ws_size;
  const float* x    = (const float*)d_in[0];   // [4096,3,28,28]
  const float* rgbw = (const float*)d_in[1];   // [100,3,5,5]
  const float* w2   = (const float*)d_in[2];   // [225,1,10,10] -> [225,100]
  const float* w3   = (const float*)d_in[3];   // [625,1,15,15] -> [625,225]
  const float* w4   = (const float*)d_in[4];   // [1225,1,25,25] -> [1225,625]
  const float* fcw  = (const float*)d_in[5];   // [10,1225]
  const float* fcb  = (const float*)d_in[6];   // [10]
  float* out = (float*)d_out;                  // [4096,10]

  // Workspace layout (floats); regions reused sequentially:
  //   R1: dist1 [4096,100,36] (59MB) -> later dist3 [12288,625]
  //   R2: dist2 [36864,225]   (33MB) -> later dist4 [4096,1225]
  //   R3: images2 [36864,100] (15MB) -> images3 [12288,225] -> images4 [4096,625]
  float* wsf      = (float*)d_ws;
  float* stats    = wsf;           // [36][2] stage1 + pairs at +72/+74/+76
  float* partials = wsf + 128;     // up to 36*1792*2 = 129024 floats
  float* R1 = wsf + 131072;
  float* R2 = R1 + (size_t)14745600;
  float* R3 = R2 + (size_t)8294400;

  // ---- Stage 1: RBF conv over RGB (per-pixel std over B*O, fused partials) --
  stage1_dist_kernel<<<dim3(256, 7, 36), 32, 0, stream>>>(x, rgbw, R1, partials);
  finalize_stage1_kernel<<<1, 64, 0, stream>>>(partials, 1792, stats);
  apply1_sfm1_kernel<<<(36864 * 100 + 255) / 256, 256, 0, stream>>>(R1, stats, R3);

  // ---- Stage 2: M=36864 N=225 K=100 ----
  rbf_dist_kernel<<<dim3(36864 / 16, (225 + 31) / 32), 32, 0, stream>>>(
      R3, w2, R2, partials, 36864, 225, 100);
  finalize_pairs_kernel<<<1, 256, 0, stream>>>(partials, (36864 / 16) * 8, stats + 72);
  apply2_sfm2_kernel<<<(12288 * 225 + 255) / 256, 256, 0, stream>>>(R2, stats + 72, R3);

  // ---- Stage 3: M=12288 N=625 K=225 ----
  rbf_dist_kernel<<<dim3(12288 / 16, (625 + 31) / 32), 32, 0, stream>>>(
      R3, w3, R1, partials, 12288, 625, 225);
  finalize_pairs_kernel<<<1, 256, 0, stream>>>(partials, (12288 / 16) * 20, stats + 74);
  apply3_sfm3_kernel<<<(4096 * 625 + 255) / 256, 256, 0, stream>>>(R1, stats + 74, R3);

  // ---- Stage 4: M=4096 N=1225 K=625 ----
  rbf_dist_kernel<<<dim3(4096 / 16, (1225 + 31) / 32), 32, 0, stream>>>(
      R3, w4, R2, partials, 4096, 1225, 625);
  finalize_pairs_kernel<<<1, 256, 0, stream>>>(partials, (4096 / 16) * 39, stats + 76);
  apply4_fc_kernel<<<(4096 * 10 + 255) / 256, 256, 0, stream>>>(
      R2, stats + 76, fcw, fcb, out);
}
